// QuantLinearGPTQ_6030134083883
// MI455X (gfx1250) — matrix operand order repaired and checked by reading, not compiled, not run on hardware
//
#include <hip/hip_runtime.h>

// GPTQ 4-bit dequant + GEMM for MI455X (gfx1250), bf16 WMMA path.
// out[M,N] = x[M,K] @ (scales[g] * (nib(qweight) - (nib(qzeros)+1)))[K,N] + bias
// M=8192, K=4096, N=11008, GROUPSIZE=128.
// Block tile 128x256 (512 thr = 16 wave32, 4x4 wave grid, 32x64 per wave), BK=64,
// double-buffered dynamic LDS (96 KB), one barrier per K-tile.

typedef __attribute__((ext_vector_type(16))) __bf16 v16bf;
typedef __attribute__((ext_vector_type(8)))  float  v8f;
typedef __attribute__((ext_vector_type(2)))  __bf16 bf16x2;

#define KDIM  4096
#define NDIM  11008
#define BK    64
#define ABUF  8192            // ushorts per A buffer (128x64 bf16 = 16 KB)
#define BBUF  16384           // ushorts per B buffer (64x256 bf16 = 32 KB)
#define LDS_BYTES ((2*ABUF + 2*BBUF) * 2)   // 96 KB

// two f32 -> packed bf16x2 (v_cvt_pk_bf16_f32)
__device__ __forceinline__ unsigned pk2(float a, float b) {
    bf16x2 v;
    v.x = (__bf16)a;
    v.y = (__bf16)b;
    return __builtin_bit_cast(unsigned, v);
}

__global__ __launch_bounds__(512)
void gptq_wmma_bf16_kernel(const float*    __restrict__ x,
                           const int*      __restrict__ qweight,
                           const int*      __restrict__ qzeros,
                           const _Float16* __restrict__ scales,
                           const _Float16* __restrict__ bias,
                           float*          __restrict__ out)
{
    extern __shared__ unsigned short smem[];   // [A0|A1|B0|B1]

    const int t    = threadIdx.x;          // 512 threads = 16 wave32s
    const int lane = t & 31;
    const int wid  = t >> 5;
    const int wm   = wid & 3;              // 4 waves along M (32 rows each)
    const int wn   = wid >> 2;             // 4 waves along N (64 cols each)
    const int bn0  = blockIdx.x * 256;
    const int bm0  = blockIdx.y * 128;

    // A-stage mapping: thread -> (row, 16-k quarter of BK=64)
    const int am = t & 127;
    const int aq = t >> 7;                 // 0..3
    const float* xrow = x + (size_t)(bm0 + am) * KDIM + aq * 16;
    const int a_slot = ((am >> 4) * 2 + (aq >> 1)) * 32 + (am & 15);
    const int a_eoff = (aq & 1) * 8;       // element offset within lane slot

    // B-stage mapping: thread -> (packed-k row 0..7, 4 columns)
    const int br   = t >> 6;               // 0..7
    const int n4   = (t & 63) * 4;         // 0..252
    const int ncol = bn0 + n4;
    const int b_kb = br >> 2;
    const int b_ln = (br & 1) * 16;
    const int b_eb = ((br >> 1) & 1) * 8;
    const int zsh  = (ncol & 7) * 4;       // 0 or 16

    // ---- software-pipeline registers ----
    float4 fa[4];                          // 16 f32 of x
    int4   qw;                             // 4 packed weight columns
    float  sc[4], mz[4];                   // scale, -scale*zero per column

    auto load_tile = [&](int kt) {
        const float* xk = xrow + kt;
        #pragma unroll
        for (int i = 0; i < 4; ++i) fa[i] = ((const float4*)xk)[i];   // global_load_b128
        qw = *(const int4*)(qweight + (size_t)((kt >> 3) + br) * NDIM + ncol);
        const int g = kt >> 7;                                        // group (BK=64 < 128)
        const unsigned zz = (unsigned)qzeros[(size_t)g * (NDIM / 8) + (ncol >> 3)];
        const _Float16* sp = scales + (size_t)g * NDIM + ncol;
        #pragma unroll
        for (int c = 0; c < 4; ++c) {
            sc[c] = (float)sp[c];
            const float z = (float)(int)(((zz >> (zsh + 4 * c)) & 0xF) + 1);
            mz[c] = -sc[c] * z;
        }
    };

    auto stage = [&](int buf) {
        // ---- A: 16 f32 -> bf16 into ISA A-fragment layout ----
        {
            const float* f = (const float*)fa;   // f[i], kk = (aq&1)*16 + i
            unsigned short* s_lo = smem + buf * ABUF + a_slot * 16 + a_eoff;
            unsigned short* s_hi = s_lo + 256;   // lane+16 slot
            uint4 q;
            q.x = pk2(f[0],  f[1]);  q.y = pk2(f[2],  f[3]);
            q.z = pk2(f[4],  f[5]);  q.w = pk2(f[6],  f[7]);
            *(uint4*)s_lo = q;                                     // ds_store_b128
            q.x = pk2(f[8],  f[9]);  q.y = pk2(f[10], f[11]);
            q.z = pk2(f[12], f[13]); q.w = pk2(f[14], f[15]);
            *(uint4*)s_hi = q;
        }
        // ---- B: nibbles -> bf16, w = fma(s, nib, -s*z) ----
        {
            unsigned short* bbase = smem + 2 * ABUF + buf * BBUF;
            const int qc[4] = { qw.x, qw.y, qw.z, qw.w };
            #pragma unroll
            for (int c = 0; c < 4; ++c) {
                const int n = n4 + c;
                unsigned short* dst =
                    bbase + (((b_kb * 16 + (n >> 4)) * 32) + b_ln + (n & 15)) * 16 + b_eb;
                const unsigned q  = (unsigned)qc[c];
                const unsigned lo = q & 0x0F0F0F0Fu;          // nibbles 0,2,4,6
                const unsigned hi = (q >> 4) & 0x0F0F0F0Fu;   // nibbles 1,3,5,7
                const float s = sc[c], m = mz[c];
                uint4 u;
                float fe, fo;                                  // v_cvt_f32_ubyte{b}
                fe = (float)( lo        & 0xFF); fo = (float)( hi        & 0xFF);
                u.x = pk2(__builtin_fmaf(s, fe, m), __builtin_fmaf(s, fo, m));
                fe = (float)((lo >>  8) & 0xFF); fo = (float)((hi >>  8) & 0xFF);
                u.y = pk2(__builtin_fmaf(s, fe, m), __builtin_fmaf(s, fo, m));
                fe = (float)((lo >> 16) & 0xFF); fo = (float)((hi >> 16) & 0xFF);
                u.z = pk2(__builtin_fmaf(s, fe, m), __builtin_fmaf(s, fo, m));
                fe = (float)( lo >> 24        ); fo = (float)( hi >> 24        );
                u.w = pk2(__builtin_fmaf(s, fe, m), __builtin_fmaf(s, fo, m));
                *(uint4*)dst = u;                              // ds_store_b128
            }
        }
    };

    v8f acc[2][4] = {};

    // ---- prologue ----
    load_tile(0);
    stage(0);

    // ---- main pipeline: 1 barrier/tile; staging stores overlap WMMA ----
    for (int kt = 0; kt < KDIM; kt += BK) {
        const int  cur  = (kt >> 6) & 1;
        const bool more = (kt + BK) < KDIM;

        if (more) {
            __builtin_prefetch(xrow + kt + 2 * BK, 0, 0);      // global_prefetch_b8
            __builtin_prefetch(qweight + (size_t)((kt + 2 * BK) >> 3) * NDIM + ncol, 0, 0);
            load_tile(kt + BK);
        }

        __syncthreads();       // prior buffer reads done; current buffer writes visible

        if (more) stage(cur ^ 1);   // ds_stores drain while WMMAs below execute

        const unsigned short* Ab = smem + cur * ABUF;
        const unsigned short* Bb = smem + 2 * ABUF + cur * BBUF;
        #pragma unroll
        for (int kb = 0; kb < 2; ++kb) {
            v16bf a0 = *(const v16bf*)&Ab[(((wm * 2 + 0) * 2 + kb) * 32 + lane) * 16];
            v16bf a1 = *(const v16bf*)&Ab[(((wm * 2 + 1) * 2 + kb) * 32 + lane) * 16];
            v16bf b0 = *(const v16bf*)&Bb[((kb * 16 + wn * 4 + 0) * 32 + lane) * 16];
            v16bf b1 = *(const v16bf*)&Bb[((kb * 16 + wn * 4 + 1) * 32 + lane) * 16];
            v16bf b2 = *(const v16bf*)&Bb[((kb * 16 + wn * 4 + 2) * 32 + lane) * 16];
            v16bf b3 = *(const v16bf*)&Bb[((kb * 16 + wn * 4 + 3) * 32 + lane) * 16];

            acc[0][0] = __builtin_amdgcn_wmma_f32_16x16x32_bf16(false, a0, false, b0, (short)0, acc[0][0], false, false);
            acc[0][1] = __builtin_amdgcn_wmma_f32_16x16x32_bf16(false, a0, false, b1, (short)0, acc[0][1], false, false);
            acc[0][2] = __builtin_amdgcn_wmma_f32_16x16x32_bf16(false, a0, false, b2, (short)0, acc[0][2], false, false);
            acc[0][3] = __builtin_amdgcn_wmma_f32_16x16x32_bf16(false, a0, false, b3, (short)0, acc[0][3], false, false);
            acc[1][0] = __builtin_amdgcn_wmma_f32_16x16x32_bf16(false, a1, false, b0, (short)0, acc[1][0], false, false);
            acc[1][1] = __builtin_amdgcn_wmma_f32_16x16x32_bf16(false, a1, false, b1, (short)0, acc[1][1], false, false);
            acc[1][2] = __builtin_amdgcn_wmma_f32_16x16x32_bf16(false, a1, false, b2, (short)0, acc[1][2], false, false);
            acc[1][3] = __builtin_amdgcn_wmma_f32_16x16x32_bf16(false, a1, false, b3, (short)0, acc[1][3], false, false);
        }
    }

    // ---- epilogue: C layout (VGPR r: lanes<16 -> M=r, lanes>=16 -> M=r+8) ----
    const int mbase = bm0 + wm * 32 + ((lane >> 4) << 3);
    #pragma unroll
    for (int mi = 0; mi < 2; ++mi) {
        #pragma unroll
        for (int ni = 0; ni < 4; ++ni) {
            const int nc = bn0 + wn * 64 + ni * 16 + (lane & 15);
            const float bv = (float)bias[nc];
            #pragma unroll
            for (int r = 0; r < 8; ++r) {
                const int m = mbase + mi * 16 + r;
                out[(size_t)m * NDIM + nc] = acc[mi][ni][r] + bv;
            }
        }
    }
}

extern "C" void kernel_launch(void* const* d_in, const int* in_sizes, int n_in,
                              void* d_out, int out_size, void* d_ws, size_t ws_size,
                              hipStream_t stream) {
    const float*    x       = (const float*)d_in[0];
    const int*      qweight = (const int*)d_in[1];
    const int*      qzeros  = (const int*)d_in[2];
    const _Float16* scales  = (const _Float16*)d_in[3];
    // d_in[4] = g_idx: always k/128 for this config; folded into kernel as kt>>7.
    const _Float16* bias    = (const _Float16*)d_in[5];
    float*          out     = (float*)d_out;

    // allow 96 KB dynamic LDS (WGP has 320 KB)
    (void)hipFuncSetAttribute((const void*)gptq_wmma_bf16_kernel,
                              hipFuncAttributeMaxDynamicSharedMemorySize, LDS_BYTES);

    const int M = in_sizes[0] / KDIM;          // 8192
    dim3 grid(NDIM / 256, M / 128);            // 43 x 64 tiles
    gptq_wmma_bf16_kernel<<<grid, 512, LDS_BYTES, stream>>>(x, qweight, qzeros, scales, bias, out);
}
